// Decoder_83597243449409
// MI455X (gfx1250) — compile-verified, matrix operand order checked by reading, not run on hardware
//
#include <hip/hip_runtime.h>
#include <cstddef>
#include <cstdint>

typedef __attribute__((ext_vector_type(16))) _Float16 v16h;
typedef __attribute__((ext_vector_type(8)))  float    v8f;

#define WMMA_F16(a, b, c) \
  __builtin_amdgcn_wmma_f32_16x16x32_f16(false, (a), false, (b), (short)0, (c), false, false)

// Load a 16-bit WMMA fragment (A: 16xK rows, or B: NxK "transposed" rows) from a
// row-major [rows][ld] f16 buffer. CDNA5 16-bit A layout: lanes 0-15 hold rows,
// halves 0..7 = K kb..kb+7, halves 8..15 = K kb+16..kb+23, kb = (lane>=16)*8.
__device__ __forceinline__ v16h ld_frag(const _Float16* base, int ld, int r0, int k0, int lane) {
  const _Float16* p = base + (size_t)(r0 + (lane & 15)) * (size_t)ld
                           + (size_t)(k0 + ((lane >> 4) << 3));
  v16h f;
#pragma unroll
  for (int i = 0; i < 8; ++i) f[i] = p[i];
#pragma unroll
  for (int i = 0; i < 8; ++i) f[i + 8] = p[16 + i];
  return f;
}

__device__ __forceinline__ float sigmf(float x) { return 1.f / (1.f + expf(-x)); }

// ---------------- async A-panel staging (GLOBAL_LOAD_ASYNC_TO_LDS, ASYNCcnt) ----
// Copies one 16-row x len-half chunk of A into LDS (stride 128 halves) using the
// CDNA5 async-to-LDS engine: 256 threads, each moves 16B (one b128).
__device__ __forceinline__ void async_copy_panel(const _Float16* A, _Float16* sdst,
                                                 int Kp, int m0, int kc, int len, int tid) {
  int row = tid >> 4;           // 0..15
  int col = (tid & 15) << 3;    // half offset within chunk, 0..120
  if (col < len) {
    unsigned lds = (unsigned)(uintptr_t)(sdst + row * 128 + col);  // LDS aperture: addr[31:0]
    unsigned long long ga =
        (unsigned long long)(uintptr_t)(A + (size_t)(m0 + row) * Kp + kc + col);
    asm volatile("global_load_async_to_lds_b128 %0, %1, off"
                 :: "v"(lds), "v"(ga) : "memory");
  }
}

// Double-buffered staged GEMM tile: A panel shared by the whole block via LDS,
// B streamed per-wave straight from L2. All threads must call (barriers inside).
__device__ __forceinline__ v8f staged_tile(const _Float16* A, const _Float16* Bw,
                                           int Kp, int m0, int n0, _Float16* sA,
                                           int tid, int lane, bool active) {
  v8f acc = {};
  int nch = (Kp + 127) >> 7;
  int len0 = (Kp < 128) ? Kp : 128;
  async_copy_panel(A, sA, Kp, m0, 0, len0, tid);
  for (int c = 0; c < nch; ++c) {
    int kc = c << 7;
    int len = Kp - kc; if (len > 128) len = 128;
    int nxt = kc + 128;
    if (nxt < Kp) {
      int nlen = Kp - nxt; if (nlen > 128) nlen = 128;
      async_copy_panel(A, sA + ((c + 1) & 1) * (16 * 128), Kp, m0, nxt, nlen, tid);
      asm volatile("s_wait_asynccnt 1" ::: "memory");  // chunk c landed
    } else {
      asm volatile("s_wait_asynccnt 0" ::: "memory");
    }
    __syncthreads();
    if (active) {
      const _Float16* cb = sA + (c & 1) * (16 * 128);
      if (len == 128) {
        // fast path: fully unrolled so the compiler can clause all 16 loads and
        // stagger the waits across 4 back-to-back WMMAs
#pragma unroll
        for (int k = 0; k < 128; k += 32)
          acc = WMMA_F16(ld_frag(cb, 128, 0, k, lane),
                         ld_frag(Bw, Kp, n0, kc + k, lane), acc);
      } else {
#pragma unroll 1
        for (int k = 0; k < len; k += 32)
          acc = WMMA_F16(ld_frag(cb, 128, 0, k, lane),
                         ld_frag(Bw, Kp, n0, kc + k, lane), acc);
      }
    }
    __syncthreads();  // buffer safe for next prefetch
  }
  return acc;
}

// ---------------------------------------------------------------- converters
__global__ void k_cvt(const float* __restrict__ src, _Float16* __restrict__ dst,
                      int rows, int src_rl, int dst_rl) {
  size_t total = (size_t)rows * dst_rl;
  for (size_t i = (size_t)blockIdx.x * 256 + threadIdx.x; i < total;
       i += (size_t)gridDim.x * 256) {
    int k = (int)(i % dst_rl);
    size_t r = i / dst_rl;
    dst[i] = (k < src_rl) ? (_Float16)src[r * src_rl + k] : (_Float16)0.f;
  }
}

// src f32 [K][N] -> dst f16 [N][Kp], zero-padded K..Kp
__global__ void k_cvtT(const float* __restrict__ src, _Float16* __restrict__ dst,
                       int K, int N, int Kp) {
  size_t total = (size_t)N * Kp;
  for (size_t i = (size_t)blockIdx.x * 256 + threadIdx.x; i < total;
       i += (size_t)gridDim.x * 256) {
    int k = (int)(i % Kp);
    int n = (int)(i / Kp);
    dst[i] = (k < K) ? (_Float16)src[(size_t)k * N + n] : (_Float16)0.f;
  }
}

// prenet input: prev[b][l] = (l==0 ? 0 : ys[b][l-1]), K padded 80 -> 96
__global__ void k_build_prev(const float* __restrict__ ys, _Float16* __restrict__ dst) {
  size_t total = (size_t)4000 * 96;
  for (size_t i = (size_t)blockIdx.x * 256 + threadIdx.x; i < total;
       i += (size_t)gridDim.x * 256) {
    int k = (int)(i % 96);
    int row = (int)(i / 96);
    int b = row / 250, l = row - b * 250;
    float v = 0.f;
    if (k < 80 && l > 0) v = ys[((size_t)(b * 250 + l - 1)) * 80 + k];
    dst[i] = (_Float16)v;
  }
}

// im2col for K=5 pad=2 conv over [16][Cin][250] -> A[(b,t)][ci*5+k], K padded to Kp
template <typename T>
__global__ void k_im2col(const T* __restrict__ x, _Float16* __restrict__ A, int Cin, int Kp) {
  size_t total = (size_t)4000 * Kp;
  int CK = Cin * 5;
  for (size_t i = (size_t)blockIdx.x * 256 + threadIdx.x; i < total;
       i += (size_t)gridDim.x * 256) {
    int kk = (int)(i % Kp);
    int row = (int)(i / Kp);
    _Float16 v = (_Float16)0.f;
    if (kk < CK) {
      int ci = kk / 5, k = kk - ci * 5;
      int b = row / 250, t = row - b * 250;
      int tt = t + k - 2;
      if (tt >= 0 && tt < 250) v = (_Float16)(float)x[((size_t)(b * Cin + ci)) * 250 + tt];
    }
    A[i] = v;
  }
}

// ---------------------------------------------------------------- WMMA GEMMs
// Block = 8 waves sharing one 16-row A panel (async-staged in LDS); wave w owns
// n-tile ng*8+w. Grid = mTiles * nGroups.
__global__ __launch_bounds__(256) void k_gemm_bias32(
    const _Float16* __restrict__ A, const _Float16* __restrict__ Bw,
    const float* __restrict__ bias, float* __restrict__ C,
    int N, int Kp, int nTn, int nGroups) {
  __shared__ _Float16 sA[2 * 16 * 128];
  int tid = threadIdx.x, wv = tid >> 5, lane = tid & 31;
  int mTile = blockIdx.x / nGroups, ng = blockIdx.x - mTile * nGroups;
  int myN = ng * 8 + wv;
  bool active = myN < nTn;
  int m0 = mTile * 16, n0 = myN * 16;
  v8f acc = staged_tile(A, Bw, Kp, m0, n0, sA, tid, lane, active);
  if (!active) return;
  int n = n0 + (lane & 15), mb = m0 + ((lane >> 4) << 3);
  float bv = bias ? bias[n] : 0.f;
#pragma unroll
  for (int r = 0; r < 8; ++r) C[(size_t)(mb + r) * N + n] = acc[r] + bv;
}

__global__ __launch_bounds__(256) void k_gemm_relu16(
    const _Float16* __restrict__ A, const _Float16* __restrict__ Bw,
    _Float16* __restrict__ C, int N, int Kp, int nTn, int nGroups) {
  __shared__ _Float16 sA[2 * 16 * 128];
  int tid = threadIdx.x, wv = tid >> 5, lane = tid & 31;
  int mTile = blockIdx.x / nGroups, ng = blockIdx.x - mTile * nGroups;
  int myN = ng * 8 + wv;
  bool active = myN < nTn;
  int m0 = mTile * 16, n0 = myN * 16;
  v8f acc = staged_tile(A, Bw, Kp, m0, n0, sA, tid, lane, active);
  if (!active) return;
  int n = n0 + (lane & 15), mb = m0 + ((lane >> 4) << 3);
#pragma unroll
  for (int r = 0; r < 8; ++r) C[(size_t)(mb + r) * N + n] = (_Float16)fmaxf(acc[r], 0.f);
}

// postnet GEMM: rows are (b,t); finalFlag=0 -> Xout[b][co][t] = tanh(g*acc+b) (f16)
// finalFlag=1 -> Fout[(b*250+t)*80+co] = g*acc+b + outs_ref[same] (f32)
__global__ __launch_bounds__(256) void k_gemm_post(
    const _Float16* __restrict__ A, const _Float16* __restrict__ Bw,
    const float* __restrict__ gam, const float* __restrict__ bet,
    _Float16* __restrict__ Xout, const float* __restrict__ outs_ref,
    float* __restrict__ Fout, int N, int Kp, int nTn, int nGroups, int finalFlag) {
  __shared__ _Float16 sA[2 * 16 * 128];
  int tid = threadIdx.x, wv = tid >> 5, lane = tid & 31;
  int mTile = blockIdx.x / nGroups, ng = blockIdx.x - mTile * nGroups;
  int myN = ng * 8 + wv;
  bool active = myN < nTn;
  int m0 = mTile * 16, n0 = myN * 16;
  v8f acc = staged_tile(A, Bw, Kp, m0, n0, sA, tid, lane, active);
  if (!active) return;
  int n = n0 + (lane & 15), mb = m0 + ((lane >> 4) << 3);
  float g = gam[n], be = bet[n];
#pragma unroll
  for (int r = 0; r < 8; ++r) {
    int row = mb + r;
    int b = row / 250, t = row - b * 250;
    float v = g * acc[r] + be;
    if (finalFlag) {
      size_t o = (size_t)row * 80 + n;
      Fout[o] = v + outs_ref[o];
    } else {
      Xout[((size_t)(b * N + n)) * 250 + t] = (_Float16)tanhf(v);
    }
  }
}

// ------------------------------------------------------- persistent decoder
__device__ __forceinline__ void lstm_step(
    const _Float16* Ain, int Ka, const _Float16* Hin,
    const _Float16* Wih, const _Float16* Whh, const float* bias, float* cg,
    _Float16* zoutA, _Float16* zoutH, int wv, int lane) {
  float z[2][8];
  for (int q = 0; q < 2; ++q) {
    int u0 = (wv * 2 + q) * 16;
    v8f g[4];
#pragma unroll
    for (int gi = 0; gi < 4; ++gi) {
      int n0 = gi * 1024 + u0;
      v8f acc = {};
      for (int k = 0; k < Ka; k += 32)
        acc = WMMA_F16(ld_frag(Ain, 1024, 0, k, lane), ld_frag(Wih, Ka, n0, k, lane), acc);
      for (int k = 0; k < 1024; k += 32)
        acc = WMMA_F16(ld_frag(Hin, 1024, 0, k, lane), ld_frag(Whh, 1024, n0, k, lane), acc);
      g[gi] = acc;
    }
    int u = u0 + (lane & 15);
    int mb = (lane >> 4) << 3;
#pragma unroll
    for (int r = 0; r < 8; ++r) {
      int m = mb + r;
      float gi_ = sigmf(g[0][r] + bias[u]);
      float gf  = sigmf(g[1][r] + bias[1024 + u]);
      float gg  = tanhf(g[2][r] + bias[2048 + u]);
      float go  = sigmf(g[3][r] + bias[3072 + u]);
      float c2 = gf * cg[m * 1024 + u] + gi_ * gg;
      cg[m * 1024 + u] = c2;
      z[q][r] = go * tanhf(c2);
    }
  }
  __syncthreads();  // everyone done reading Ain/Hin before overwrite
#pragma unroll
  for (int q = 0; q < 2; ++q) {
    int u = (wv * 2 + q) * 16 + (lane & 15);
    int mb = (lane >> 4) << 3;
#pragma unroll
    for (int r = 0; r < 8; ++r) {
      _Float16 h = (_Float16)z[q][r];
      int m = mb + r;
      if (zoutA) zoutA[m * 1024 + u] = h;
      zoutH[m * 1024 + u] = h;
    }
  }
  __syncthreads();
}

__global__ __launch_bounds__(1024) void k_decoder(
    const float* __restrict__ hs, const int* __restrict__ hlens,
    const float* __restrict__ preenc, const float* __restrict__ loc_w,
    const float* __restrict__ W_att, const float* __restrict__ gvec_w,
    const float* __restrict__ gvec_b, const _Float16* __restrict__ WdecT,
    const _Float16* __restrict__ Wih0, const _Float16* __restrict__ Whh0,
    const float* __restrict__ b0, const _Float16* __restrict__ Wih1,
    const _Float16* __restrict__ Whh1, const float* __restrict__ b1,
    const _Float16* __restrict__ featT, const float* __restrict__ prob_W,
    const float* __restrict__ prob_b, const _Float16* __restrict__ pren,
    float* __restrict__ c0g, float* __restrict__ c1g,
    float* __restrict__ outs_nct, float* __restrict__ outs_blo,
    float* __restrict__ probs) {
  __shared__ _Float16 bufA[16 * 1024];   // LSTM input x / z0 (f16 A-operand)
  __shared__ _Float16 bufH0[16 * 1024];  // hidden z0
  __shared__ _Float16 bufH1[16 * 1024];  // hidden z1
  __shared__ float attw[16 * 200];       // attention weights
  __shared__ float ev[16 * 200];         // energies / exp scratch
  __shared__ float decp[16 * 128];       // z0 @ W_dec

  int tid = threadIdx.x, lane = tid & 31, wv = tid >> 5;

  for (int i = tid; i < 16 * 1024; i += 1024) {
    bufH0[i] = (_Float16)0.f;
    bufH1[i] = (_Float16)0.f;
    c0g[i] = 0.f;
    c1g[i] = 0.f;
  }
  for (int i = tid; i < 3200; i += 1024) {
    int b = i / 200, t = i - b * 200;
    int hl = hlens[b];
    attw[i] = (t < hl) ? (1.f / (float)hl) : 0.f;
  }
  __builtin_prefetch(Wih0, 0, 1);
  __builtin_prefetch(Whh0, 0, 1);
  __builtin_prefetch(Wih1, 0, 1);
  __builtin_prefetch(Whh1, 0, 1);
  __syncthreads();

  for (int step = 0; step < 250; ++step) {
    // ---- decoder-state projection: decp = z0 @ W_dec (16x128) via WMMA ----
    if (wv < 8) {
      int n0 = wv * 16;
      v8f acc = {};
      for (int k = 0; k < 1024; k += 32)
        acc = WMMA_F16(ld_frag(bufH0, 1024, 0, k, lane),
                       ld_frag(WdecT, 1024, n0, k, lane), acc);
      int n = n0 + (lane & 15), mb = (lane >> 4) << 3;
#pragma unroll
      for (int r = 0; r < 8; ++r) decp[(mb + r) * 128 + n] = acc[r];
    }
    __syncthreads();

    // ---- location conv + attention energies ----
    for (int pos = tid; pos < 3200; pos += 1024) {
      int b = pos / 200, t = pos - b * 200;
      float cs[32];
#pragma unroll 1
      for (int c = 0; c < 32; ++c) {
        float s = 0.f;
        for (int k = 0; k < 31; ++k) {
          int tt = t + k - 15;
          if (tt >= 0 && tt < 200) s += attw[b * 200 + tt] * loc_w[c * 31 + k];
        }
        cs[c] = s;
      }
      float ea = 0.f;
      const float* pe = preenc + (size_t)pos * 128;
#pragma unroll 1
      for (int a = 0; a < 128; ++a) {
        float v = pe[a] + decp[b * 128 + a];
#pragma unroll
        for (int c = 0; c < 32; ++c) v += cs[c] * W_att[c * 128 + a];
        ea += gvec_w[a] * tanhf(v);
      }
      float e = 2.f * (ea + gvec_b[0]);
      ev[pos] = (t < hlens[b]) ? e : -2.0e30f;
    }
    __syncthreads();

    // ---- softmax over T, one wave32 per batch row ----
    if (wv < 16) {
      int b = wv;
      float mx = -3.0e38f;
      for (int t = lane; t < 200; t += 32) mx = fmaxf(mx, ev[b * 200 + t]);
      for (int o = 16; o; o >>= 1) mx = fmaxf(mx, __shfl_xor(mx, o, 32));
      float sm = 0.f;
      for (int t = lane; t < 200; t += 32) {
        float ex = expf(ev[b * 200 + t] - mx);
        ev[b * 200 + t] = ex;
        sm += ex;
      }
      for (int o = 16; o; o >>= 1) sm += __shfl_xor(sm, o, 32);
      float inv = 1.f / sm;
      for (int t = lane; t < 200; t += 32) attw[b * 200 + t] = ev[b * 200 + t] * inv;
    }
    __syncthreads();

    // ---- context vector + LSTM0 input (f16 into LDS) ----
    for (int i = tid; i < 16 * 512; i += 1024) {
      int b = i >> 9, d = i & 511;
      const float* hp = hs + ((size_t)b * 200) * 512 + d;
      const float* wp = attw + b * 200;
      float s = 0.f;
      for (int t = 0; t < 200; ++t) s += wp[t] * hp[(size_t)t * 512];
      bufA[b * 1024 + d] = (_Float16)s;
    }
    for (int i = tid; i < 16 * 256; i += 1024) {
      int b = i >> 8, j = i & 255;
      bufA[b * 1024 + 512 + j] = pren[((size_t)b * 250 + step) * 256 + j];
    }
    __syncthreads();

    lstm_step(bufA, 768, bufH0, Wih0, Whh0, b0, c0g, bufA, bufH0, wv, lane);
    lstm_step(bufA, 1024, bufH1, Wih1, Whh1, b1, c1g, nullptr, bufH1, wv, lane);

    // ---- output heads: feat (WMMA 16x80) + stop-prob dot ----
    if (wv < 5) {
      int n0 = wv * 16;
      v8f acc = {};
      for (int k = 0; k < 1024; k += 32)
        acc = WMMA_F16(ld_frag(bufH1, 1024, 0, k, lane),
                       ld_frag(featT, 1024, n0, k, lane), acc);
      int n = n0 + (lane & 15), mb = (lane >> 4) << 3;
#pragma unroll
      for (int r = 0; r < 8; ++r) {
        int b = mb + r;
        float v = acc[r];
        outs_nct[((size_t)b * 80 + n) * 250 + step] = v;
        outs_blo[((size_t)b * 250 + step) * 80 + n] = v;
      }
    } else if (wv == 5) {
      for (int b = 0; b < 16; ++b) {
        float s = 0.f;
        for (int u = lane; u < 1024; u += 32) s += (float)bufH1[b * 1024 + u] * prob_W[u];
        for (int o = 16; o; o >>= 1) s += __shfl_xor(s, o, 32);
        if (lane == 0) probs[b * 250 + step] = s + prob_b[0];
      }
    }
    __syncthreads();
  }
}

// ---------------------------------------------------------------- launcher
extern "C" void kernel_launch(void* const* d_in, const int* in_sizes, int n_in,
                              void* d_out, int out_size, void* d_ws, size_t ws_size,
                              hipStream_t stream) {
  (void)in_sizes; (void)n_in; (void)out_size; (void)ws_size;
  const float* hs     = (const float*)d_in[0];
  const int*   hlens  = (const int*)d_in[1];
  const float* ys     = (const float*)d_in[2];
  const float* W_enc  = (const float*)d_in[3];
  const float* b_enc  = (const float*)d_in[4];
  const float* W_dec  = (const float*)d_in[5];
  const float* W_att  = (const float*)d_in[6];
  const float* loc_w  = (const float*)d_in[7];
  const float* gvec_w = (const float*)d_in[8];
  const float* gvec_b = (const float*)d_in[9];
  const float* pW1    = (const float*)d_in[10];
  const float* pW2    = (const float*)d_in[11];
  const float* Wih0   = (const float*)d_in[12];
  const float* Whh0   = (const float*)d_in[13];
  const float* b0     = (const float*)d_in[14];
  const float* Wih1   = (const float*)d_in[15];
  const float* Whh1   = (const float*)d_in[16];
  const float* b1     = (const float*)d_in[17];
  const float* feat_W = (const float*)d_in[18];
  const float* prob_W = (const float*)d_in[19];
  const float* prob_b = (const float*)d_in[20];
  const float* pw_in  = (const float*)d_in[21];
  const float* pg_in  = (const float*)d_in[22];
  const float* pb_in  = (const float*)d_in[23];
  const float* pw_mid = (const float*)d_in[24];
  const float* pg_mid = (const float*)d_in[25];
  const float* pb_mid = (const float*)d_in[26];
  const float* pw_out = (const float*)d_in[27];
  const float* pg_out = (const float*)d_in[28];
  const float* pb_out = (const float*)d_in[29];

  float* out_post  = (float*)d_out;            // [16,250,80]
  float* out_feats = out_post + 16 * 250 * 80; // [16,250,80]
  float* out_probs = out_feats + 16 * 250 * 80;// [16,250]

  char* bump = (char*)d_ws;
  auto alloc = [&](size_t bytes) -> char* {
    char* r = bump;
    bump += (bytes + 255) & ~(size_t)255;
    return r;
  };
  _Float16* hsH   = (_Float16*)alloc((size_t)3200 * 512 * 2);
  _Float16* WencT = (_Float16*)alloc((size_t)128 * 512 * 2);
  float*    preenc= (float*)alloc((size_t)3200 * 128 * 4);
  _Float16* prevA = (_Float16*)alloc((size_t)4000 * 96 * 2);
  _Float16* W1T   = (_Float16*)alloc((size_t)256 * 96 * 2);
  _Float16* W2T   = (_Float16*)alloc((size_t)256 * 256 * 2);
  _Float16* pm1   = (_Float16*)alloc((size_t)4000 * 256 * 2);
  _Float16* pren  = (_Float16*)alloc((size_t)4000 * 256 * 2);
  _Float16* WdecT = (_Float16*)alloc((size_t)128 * 1024 * 2);
  _Float16* Wih0H = (_Float16*)alloc((size_t)4096 * 768 * 2);
  _Float16* Whh0H = (_Float16*)alloc((size_t)4096 * 1024 * 2);
  _Float16* Wih1H = (_Float16*)alloc((size_t)4096 * 1024 * 2);
  _Float16* Whh1H = (_Float16*)alloc((size_t)4096 * 1024 * 2);
  _Float16* featT = (_Float16*)alloc((size_t)80 * 1024 * 2);
  float*    c0g   = (float*)alloc((size_t)16 * 1024 * 4);
  float*    c1g   = (float*)alloc((size_t)16 * 1024 * 4);
  float*    outs_nct = (float*)alloc((size_t)16 * 80 * 250 * 4);
  _Float16* pBin  = (_Float16*)alloc((size_t)512 * 416 * 2);
  _Float16* pBmid = (_Float16*)alloc((size_t)3 * 512 * 2560 * 2);
  _Float16* pBout = (_Float16*)alloc((size_t)80 * 2560 * 2);
  _Float16* col   = (_Float16*)alloc((size_t)4000 * 2560 * 2);
  _Float16* xb0   = (_Float16*)alloc((size_t)16 * 512 * 250 * 2);
  _Float16* xb1   = (_Float16*)alloc((size_t)16 * 512 * 250 * 2);

  dim3 gs(1024), bs(256);

  // encoder projection: pre_enc = hs @ W_enc + b_enc  (M=3200, N=128, K=512)
  k_cvt<<<gs, bs, 0, stream>>>(hs, hsH, 3200, 512, 512);
  k_cvtT<<<gs, bs, 0, stream>>>(W_enc, WencT, 512, 128, 512);
  k_gemm_bias32<<<200 * 1, 256, 0, stream>>>(hsH, WencT, b_enc, preenc, 128, 512, 8, 1);

  // prenet (vectorized over all 250 teacher-forced steps)
  k_build_prev<<<gs, bs, 0, stream>>>(ys, prevA);
  k_cvtT<<<gs, bs, 0, stream>>>(pW1, W1T, 80, 256, 96);
  k_cvtT<<<gs, bs, 0, stream>>>(pW2, W2T, 256, 256, 256);
  k_gemm_relu16<<<250 * 2, 256, 0, stream>>>(prevA, W1T, pm1, 256, 96, 16, 2);
  k_gemm_relu16<<<250 * 2, 256, 0, stream>>>(pm1, W2T, pren, 256, 256, 16, 2);

  // f16 weight images for the recurrent core
  k_cvtT<<<gs, bs, 0, stream>>>(W_dec, WdecT, 1024, 128, 1024);
  k_cvt<<<gs, bs, 0, stream>>>(Wih0, Wih0H, 4096, 768, 768);
  k_cvt<<<gs, bs, 0, stream>>>(Whh0, Whh0H, 4096, 1024, 1024);
  k_cvt<<<gs, bs, 0, stream>>>(Wih1, Wih1H, 4096, 1024, 1024);
  k_cvt<<<gs, bs, 0, stream>>>(Whh1, Whh1H, 4096, 1024, 1024);
  k_cvtT<<<gs, bs, 0, stream>>>(feat_W, featT, 1024, 80, 1024);

  // persistent sequential decoder: 1 workgroup, 32 waves, 250 steps
  k_decoder<<<1, 1024, 0, stream>>>(hs, hlens, preenc, loc_w, W_att, gvec_w, gvec_b,
                                    WdecT, Wih0H, Whh0H, b0, Wih1H, Whh1H, b1,
                                    featT, prob_W, prob_b, pren,
                                    c0g, c1g, outs_nct, out_feats, out_probs);

  // postnet weights (B matrices are already [Cout][Cin*5] row-major)
  k_cvt<<<gs, bs, 0, stream>>>(pw_in, pBin, 512, 400, 416);
  k_cvt<<<gs, bs, 0, stream>>>(pw_mid, pBmid, 1536, 2560, 2560);
  k_cvt<<<gs, bs, 0, stream>>>(pw_out, pBout, 80, 2560, 2560);

  // postnet as im2col + WMMA GEMMs (M=4000 rows of (b,t))
  k_im2col<float><<<gs, bs, 0, stream>>>(outs_nct, col, 80, 416);
  k_gemm_post<<<250 * 4, 256, 0, stream>>>(col, pBin, pg_in, pb_in, xb0,
                                           nullptr, nullptr, 512, 416, 32, 4, 0);
  _Float16* xc = xb0; _Float16* xn = xb1;
  for (int i = 0; i < 3; ++i) {
    k_im2col<_Float16><<<gs, bs, 0, stream>>>(xc, col, 512, 2560);
    k_gemm_post<<<250 * 4, 256, 0, stream>>>(col, pBmid + (size_t)i * 512 * 2560,
                                             pg_mid + i * 512, pb_mid + i * 512, xn,
                                             nullptr, nullptr, 512, 2560, 32, 4, 0);
    _Float16* tmp = xc; xc = xn; xn = tmp;
  }
  k_im2col<_Float16><<<gs, bs, 0, stream>>>(xc, col, 512, 2560);
  k_gemm_post<<<250 * 1, 256, 0, stream>>>(col, pBout, pg_out, pb_out, nullptr,
                                           out_feats, out_post, 80, 2560, 5, 1, 1);
}
